// ExpanderLinear_15487652069503
// MI455X (gfx1250) — compile-verified, matrix operand order checked by reading, not run on hardware
//
#include <hip/hip_runtime.h>

// CDNA5 (gfx1250) ExpanderLinear: Y = X @ (W .* mask)^T + bias
// bf16 WMMA (v_wmma_f32_16x16x32_bf16), f32 accumulate, double-buffered LDS.

typedef __attribute__((ext_vector_type(16))) __bf16 v16bf;
typedef __attribute__((ext_vector_type(8)))  float  v8f;

#define TOKENS 8192
#define INDIM  4096
#define OUTDIM 4096

#define TM 128            // output rows per workgroup
#define TN 128            // output cols per workgroup
#define TK 64             // k-slice per LDS stage
#define LDT (TK + 8)      // padded LDS row stride in bf16 elems (144 B, 16B-aligned)

union FragU { v16bf v; uint4 u[2]; };

// pack two float4 into 8 bf16 (RNE via clang __bf16 cast)
__device__ inline uint4 pack8(float4 a, float4 b) {
    union { uint4 u; __bf16 h[8]; } r;
    r.h[0] = (__bf16)a.x; r.h[1] = (__bf16)a.y; r.h[2] = (__bf16)a.z; r.h[3] = (__bf16)a.w;
    r.h[4] = (__bf16)b.x; r.h[5] = (__bf16)b.y; r.h[6] = (__bf16)b.z; r.h[7] = (__bf16)b.w;
    return r.u;
}

__device__ inline float4 apply_mask(float4 w, int4 m) {
    float4 r;
    r.x = m.x ? w.x : 0.0f;
    r.y = m.y ? w.y : 0.0f;
    r.z = m.z ? w.z : 0.0f;
    r.w = m.w ? w.w : 0.0f;
    return r;
}

// global load (128-bit), fuse mask, convert to bf16; keeps only 8 uint4 live
__device__ inline void stage_regs(const float* __restrict__ xP,
                                  const float* __restrict__ wP,
                                  const int*   __restrict__ mP,
                                  uint4 aP[4], uint4 bP[4]) {
#pragma unroll
    for (int q = 0; q < 8; q += 2) {
        float4 x0 = ((const float4*)xP)[q];
        float4 x1 = ((const float4*)xP)[q + 1];
        aP[q >> 1] = pack8(x0, x1);
        float4 w0 = ((const float4*)wP)[q];
        float4 w1 = ((const float4*)wP)[q + 1];
        int4   m0 = ((const int4*)mP)[q];
        int4   m1 = ((const int4*)mP)[q + 1];
        bP[q >> 1] = pack8(apply_mask(w0, m0), apply_mask(w1, m1));
    }
}

__device__ inline void stage_store(__bf16* sAb, __bf16* sBb, int off,
                                   const uint4 aP[4], const uint4 bP[4]) {
    uint4* dA = (uint4*)(sAb + off);
    uint4* dB = (uint4*)(sBb + off);
#pragma unroll
    for (int c = 0; c < 4; ++c) { dA[c] = aP[c]; dB[c] = bP[c]; }
}

__global__ __launch_bounds__(256)
void expander_linear_wmma(const float* __restrict__ X,
                          const float* __restrict__ W,
                          const float* __restrict__ bias,
                          const int*   __restrict__ Mk,
                          float* __restrict__ Y)
{
    __shared__ __bf16 sA[2][TM * LDT];   // X tiles (bf16, row-major, double buffer)
    __shared__ __bf16 sB[2][TN * LDT];   // masked W tiles

    const int tid  = threadIdx.x;
    const int lane = tid & 31;
    const int wave = tid >> 5;        // 0..7
    const int wm   = wave >> 1;       // 0..3  -> 32-row slab
    const int wn   = wave & 1;        // 0..1  -> 64-col slab
    const int half = lane >> 4;       // hi/lo 16 lanes
    const int l16  = lane & 15;

    const int blockM = blockIdx.y * TM;
    const int blockN = blockIdx.x * TN;

    // staging assignment: 2 threads per tile row, 32 f32 elems each
    const int ldRow = tid >> 1;            // 0..127
    const int ldCol = (tid & 1) * 32;      // 0 or 32
    const int stoff = ldRow * LDT + ldCol; // LDS store offset (16B-aligned)

    const float* xRow = X  + (size_t)(blockM + ldRow) * INDIM + ldCol;
    const float* wRow = W  + (size_t)(blockN + ldRow) * INDIM + ldCol;
    const int*   mRow = Mk + (size_t)(blockN + ldRow) * INDIM + ldCol;

    v8f acc[2][4] = {};   // 8 WMMA accumulators (32x64 slab per wave)

    const int NT = INDIM / TK;   // 64 k-tiles

    // ---- prologue: stage tile 0 ----
    uint4 aP[4], bP[4];
    stage_regs(xRow, wRow, mRow, aP, bP);
    stage_store(sA[0], sB[0], stoff, aP, bP);
    __syncthreads();

    for (int t = 0; t < NT; ++t) {
        const bool more = (t + 1) < NT;
        if (more) {
            const int kn = (t + 1) * TK;
            stage_regs(xRow + kn, wRow + kn, mRow + kn, aP, bP);
            if (t + 2 < NT) {   // global_prefetch_b8, two tiles ahead
                __builtin_prefetch(xRow + kn + TK, 0, 1);
                __builtin_prefetch(wRow + kn + TK, 0, 1);
                __builtin_prefetch(mRow + kn + TK, 0, 1);
            }
        }

        // ---- compute on current buffer: 2 k-steps of 32, 8 WMMAs each ----
        const __bf16* cA = sA[t & 1];
        const __bf16* cB = sB[t & 1];
#pragma unroll
        for (int kk = 0; kk < TK; kk += 32) {
            FragU af[2], bg[4];
#pragma unroll
            for (int i = 0; i < 2; ++i) {
                // 16-bit A layout: lane m(=l16), half h: K = {h*8..+7, 16+h*8..+7}
                const __bf16* p = &cA[(wm * 32 + i * 16 + l16) * LDT + kk + half * 8];
                af[i].u[0] = *(const uint4*)p;
                af[i].u[1] = *(const uint4*)(p + 16);
            }
#pragma unroll
            for (int j = 0; j < 4; ++j) {
                // NT gemm: B fragment has identical per-lane K pattern (lane = n)
                const __bf16* p = &cB[(wn * 64 + j * 16 + l16) * LDT + kk + half * 8];
                bg[j].u[0] = *(const uint4*)p;
                bg[j].u[1] = *(const uint4*)(p + 16);
            }
#pragma unroll
            for (int i = 0; i < 2; ++i)
#pragma unroll
                for (int j = 0; j < 4; ++j)
                    acc[i][j] = __builtin_amdgcn_wmma_f32_16x16x32_bf16(
                        /*neg_a=*/false, af[i].v,
                        /*neg_b=*/false, bg[j].v,
                        /*c_mod=*/(short)0, acc[i][j],
                        /*reuse_a=*/false, /*reuse_b=*/false);
        }

        // ---- stage tile t+1 into the other buffer (readers of it finished at
        //      the barrier ending iteration t-1), then one barrier ----
        if (more) stage_store(sA[(t + 1) & 1], sB[(t + 1) & 1], stoff, aP, bP);
        __syncthreads();
    }

    // ---- epilogue: bias add + f32 store (C/D layout: VGPR r -> M=r / 8+r) ----
#pragma unroll
    for (int j = 0; j < 4; ++j) {
        const int col = blockN + wn * 64 + j * 16 + l16;
        const float bj = bias[col];
#pragma unroll
        for (int i = 0; i < 2; ++i) {
            const int row0 = blockM + wm * 32 + i * 16 + half * 8;
#pragma unroll
            for (int r = 0; r < 8; ++r) {
                Y[(size_t)(row0 + r) * OUTDIM + col] = acc[i][j][r] + bj;
            }
        }
    }
}

extern "C" void kernel_launch(void* const* d_in, const int* in_sizes, int n_in,
                              void* d_out, int out_size, void* d_ws, size_t ws_size,
                              hipStream_t stream) {
    (void)in_sizes; (void)n_in; (void)out_size; (void)d_ws; (void)ws_size;
    const float* X  = (const float*)d_in[0];
    const float* W  = (const float*)d_in[1];
    const float* B  = (const float*)d_in[2];
    const int*   Mk = (const int*)  d_in[3];
    float* Y = (float*)d_out;

    dim3 grid(OUTDIM / TN, TOKENS / TM);   // x = N-tiles inner: X band stays L2-hot
    expander_linear_wmma<<<grid, dim3(256), 0, stream>>>(X, W, B, Mk, Y);
}
